// MultiHeadAttentionBlock_5978594476398
// MI455X (gfx1250) — compile-verified, matrix operand order checked by reading, not run on hardware
//
#include <hip/hip_runtime.h>
#include <hip/hip_fp16.h>

typedef __attribute__((ext_vector_type(16))) _Float16 v16h;
typedef __attribute__((ext_vector_type(8)))  _Float16 v8h;
typedef __attribute__((ext_vector_type(8)))  float    v8f;
typedef __attribute__((ext_vector_type(4)))  float    v4f;

#define EMBED    1024
#define HEADS    16
#define HEAD_DIM 64
#define BATCH    4
#define SEQ      2048
#define MROWS    (BATCH * SEQ)   // 8192
#define WPAD     40              // padded LDS row (halves): stride 20 dwords -> 16 distinct banks

static __device__ __forceinline__ v8f wmma16(v16h a, v16h b, v8f c) {
    return __builtin_amdgcn_wmma_f32_16x16x32_f16(
        /*neg_a=*/false, a, /*neg_b=*/false, b,
        /*c_mod=*/(short)0, c, /*reuse_a=*/false, /*reuse_b=*/false);
}

// ---------------------------------------------------------------------------
// fp32 -> f16 elementwise (weights)
// ---------------------------------------------------------------------------
__global__ void cvt_f32_f16(const float* __restrict__ src,
                            _Float16* __restrict__ dst, int n) {
    int i = blockIdx.x * blockDim.x + threadIdx.x;
    if (i < n) dst[i] = (_Float16)src[i];
}

// ---------------------------------------------------------------------------
// P = X @ W^T (NT GEMM). X fp32 [MROWS,EMBED], W f16 [EMBED,EMBED] row-major.
// Block = 8 waves sharing one 64-col W panel staged to LDS via
// global_load_async_to_lds_b128; each wave owns a 16x64 output tile.
// mode 0: write f16 head-split [B,H,S,D]; mode 1: transposed [B,H,D,S]
// ---------------------------------------------------------------------------
__global__ void proj_gemm(const float* __restrict__ X,
                          const _Float16* __restrict__ W,
                          _Float16* __restrict__ Y, int mode) {
    __shared__ _Float16 wtile[64 * WPAD];
    const int tid    = threadIdx.x;
    const int lane   = tid & 31;
    const int wslot  = tid >> 5;
    const int npanel = blockIdx.x & 15;    // 16 panels of 64 cols
    const int mgroup = blockIdx.x >> 4;    // 64 groups of 8 m-tiles
    const int n0 = npanel * 64;
    const int m0 = (mgroup * 8 + wslot) * 16;

    const int lm    = lane & 15;           // A row / B col within tile
    const int hk    = (lane >> 4) * 8;     // per-half K chunk offset
    const int mhalf = (lane >> 4) * 8;     // C/D row offset of this half

    // async staging: 256 threads x 16B = one 64x32-half W panel slice
    const int srow   = tid >> 2;           // 0..63
    const int schunk = (tid & 3) * 8;      // halves within row
    const _Float16* wsrc = W + (size_t)(n0 + srow) * EMBED + schunk;
    const unsigned  sdst = (unsigned)(uintptr_t)&wtile[srow * WPAD + schunk];

    v8f acc[4] = {};
    const float* xrow = X + (size_t)(m0 + lm) * EMBED;

    for (int k = 0; k < EMBED; k += 32) {
        __syncthreads();                   // previous panel fully consumed
        asm volatile("global_load_async_to_lds_b128 %0, %1, off"
                     :: "v"(sdst), "v"((const void*)(wsrc + k)) : "memory");
        asm volatile("s_wait_asynccnt 0" ::: "memory");
        __syncthreads();                   // panel visible to all waves

        __builtin_prefetch(xrow + k + 64, 0, 0);
        // A operand: fp32 rows -> f16, ISA 16-bit A layout
        v16h a;
        {
            const float* p = xrow + k + hk;
            v4f f0 = *(const v4f*)(p);
            v4f f1 = *(const v4f*)(p + 4);
            v4f f2 = *(const v4f*)(p + 16);
            v4f f3 = *(const v4f*)(p + 20);
#pragma unroll
            for (int i = 0; i < 4; ++i) {
                a[i]      = (_Float16)f0[i];
                a[4 + i]  = (_Float16)f1[i];
                a[8 + i]  = (_Float16)f2[i];
                a[12 + i] = (_Float16)f3[i];
            }
        }
#pragma unroll
        for (int t = 0; t < 4; ++t) {
            const _Float16* wr = &wtile[(t * 16 + lm) * WPAD + hk];
            v8h c0 = *(const v8h*)(wr);
            v8h c1 = *(const v8h*)(wr + 16);
            v16h b;
#pragma unroll
            for (int i = 0; i < 8; ++i) { b[i] = c0[i]; b[8 + i] = c1[i]; }
            acc[t] = wmma16(a, b, acc[t]);
        }
    }

#pragma unroll
    for (int t = 0; t < 4; ++t)
#pragma unroll
        for (int r = 0; r < 8; ++r) {
            int mrow = m0 + r + mhalf;
            int ncol = n0 + t * 16 + lm;
            int bb = mrow >> 11, ss = mrow & 2047;   // /SEQ, %SEQ
            int h  = ncol >> 6,  d  = ncol & 63;     // /D,   %D
            size_t off = (mode == 0)
                ? ((size_t)((bb * HEADS + h) * SEQ + ss) * HEAD_DIM + d)
                : ((size_t)((bb * HEADS + h) * HEAD_DIM + d) * SEQ + ss);
            Y[off] = (_Float16)acc[t][r];
        }
}

// ---------------------------------------------------------------------------
// Flash attention: one wave per (b, h, 16-query tile); online softmax over
// 32-key blocks.  Qh/Kh: [B,H,S,D] f16.  VT: [B,H,D,S] f16.  Xo: [B,S,E] f16.
// ---------------------------------------------------------------------------
__global__ void attn_kernel(const _Float16* __restrict__ Qh,
                            const _Float16* __restrict__ Kh,
                            const _Float16* __restrict__ VT,
                            _Float16* __restrict__ Xo) {
    __shared__ _Float16 plds[8][16 * 32];   // per-wave P tile scratch
    const int lane  = threadIdx.x & 31;
    const int wslot = threadIdx.x >> 5;
    const int wid   = blockIdx.x * 8 + wslot;
    const int qt = wid & 127;               // S/16 = 128 q tiles
    const int h  = (wid >> 7) & 15;
    const int b  = wid >> 11;
    const int q0 = qt * 16;

    const int lm    = lane & 15;
    const int hk    = (lane >> 4) * 8;
    const int mhalf = (lane >> 4) * 8;

    const _Float16* Qb = Qh + (size_t)((b * HEADS + h) * SEQ) * HEAD_DIM;
    const _Float16* Kb = Kh + (size_t)((b * HEADS + h) * SEQ) * HEAD_DIM;
    const _Float16* Vb = VT + (size_t)((b * HEADS + h) * HEAD_DIM) * SEQ;

    // Q A-operands for K-dim halves 0..31 and 32..63
    v16h aq[2];
    {
        const _Float16* qrow = Qb + (size_t)(q0 + lm) * HEAD_DIM;
#pragma unroll
        for (int half = 0; half < 2; ++half) {
            v8h c0 = *(const v8h*)(qrow + 32 * half + hk);
            v8h c1 = *(const v8h*)(qrow + 32 * half + hk + 16);
#pragma unroll
            for (int i = 0; i < 8; ++i) { aq[half][i] = c0[i]; aq[half][8 + i] = c1[i]; }
        }
    }

    float row_max[8], row_sum[8];
#pragma unroll
    for (int r = 0; r < 8; ++r) { row_max[r] = -1e30f; row_sum[r] = 0.f; }
    v8f o[4] = {};

    const float scale = 0.125f;             // 1/sqrt(64)
    const int nblocks = (q0 + 47) >> 5;     // causal: keys <= q0+15

    for (int kb = 0; kb < nblocks; ++kb) {
        const int key0 = kb * 32;
        // S = Q @ K^T (two 16x16 tiles covering 32 keys)
        v8f s[2];
#pragma unroll
        for (int t = 0; t < 2; ++t) {
            const _Float16* krow = Kb + (size_t)(key0 + t * 16 + lm) * HEAD_DIM;
            v8h c0 = *(const v8h*)(krow + hk);
            v8h c1 = *(const v8h*)(krow + hk + 16);
            v8h c2 = *(const v8h*)(krow + 32 + hk);
            v8h c3 = *(const v8h*)(krow + 32 + hk + 16);
            v16h bk0, bk1;
#pragma unroll
            for (int i = 0; i < 8; ++i) {
                bk0[i] = c0[i]; bk0[8 + i] = c1[i];
                bk1[i] = c2[i]; bk1[8 + i] = c3[i];
            }
            v8f z = {};
            z = wmma16(aq[0], bk0, z);
            z = wmma16(aq[1], bk1, z);
            s[t] = z;
        }

        // causal mask + online softmax (rows striped across 16-lane halves)
#pragma unroll
        for (int r = 0; r < 8; ++r) {
            int qrow = q0 + r + mhalf;
            float s0 = s[0][r] * scale;
            float s1 = s[1][r] * scale;
            if (key0 + lm      > qrow) s0 = -1e30f;
            if (key0 + 16 + lm > qrow) s1 = -1e30f;
            float bm = fmaxf(s0, s1);
#pragma unroll
            for (int msk = 8; msk >= 1; msk >>= 1)
                bm = fmaxf(bm, __shfl_xor(bm, msk, 32));
            float nm   = fmaxf(row_max[r], bm);
            float corr = __expf(row_max[r] - nm);
            float p0 = __expf(s0 - nm);
            float p1 = __expf(s1 - nm);
            float ps = p0 + p1;
#pragma unroll
            for (int msk = 8; msk >= 1; msk >>= 1)
                ps += __shfl_xor(ps, msk, 32);
            row_sum[r] = row_sum[r] * corr + ps;
            row_max[r] = nm;
#pragma unroll
            for (int t = 0; t < 4; ++t) o[t][r] *= corr;
            int ml = r + mhalf;
            plds[wslot][ml * 32 + lm]      = (_Float16)p0;
            plds[wslot][ml * 32 + 16 + lm] = (_Float16)p1;
        }

        // reload P (C-layout -> A-layout) via LDS; same-wave DS ops are in order
        v16h ap;
        {
            const _Float16* pr = &plds[wslot][lm * 32 + hk];
            v8h c0 = *(const v8h*)(pr);
            v8h c1 = *(const v8h*)(pr + 16);
#pragma unroll
            for (int i = 0; i < 8; ++i) { ap[i] = c0[i]; ap[8 + i] = c1[i]; }
        }

        // O += P @ V   (V^T rows are contiguous along keys)
#pragma unroll
        for (int t = 0; t < 4; ++t) {
            const _Float16* vrow = Vb + (size_t)(t * 16 + lm) * SEQ + key0;
            v8h c0 = *(const v8h*)(vrow + hk);
            v8h c1 = *(const v8h*)(vrow + hk + 16);
            v16h bv;
#pragma unroll
            for (int i = 0; i < 8; ++i) { bv[i] = c0[i]; bv[8 + i] = c1[i]; }
            o[t] = wmma16(ap, bv, o[t]);
        }
    }

    // normalize and write x back merged [B,S,E] as f16 for the output GEMM
#pragma unroll
    for (int r = 0; r < 8; ++r) {
        float inv = 1.0f / row_sum[r];
        int srow = q0 + r + mhalf;
        _Float16* xr = Xo + (size_t)(b * SEQ + srow) * EMBED + h * HEAD_DIM;
#pragma unroll
        for (int t = 0; t < 4; ++t)
            xr[t * 16 + lm] = (_Float16)(o[t][r] * inv);
    }
}

// ---------------------------------------------------------------------------
// out = X16 @ Wo^T, f16 inputs, f32 output [MROWS, EMBED].
// Same LDS-staged panel structure as proj_gemm.
// ---------------------------------------------------------------------------
__global__ void out_gemm(const _Float16* __restrict__ X,
                         const _Float16* __restrict__ W,
                         float* __restrict__ Y) {
    __shared__ _Float16 wtile[64 * WPAD];
    const int tid    = threadIdx.x;
    const int lane   = tid & 31;
    const int wslot  = tid >> 5;
    const int npanel = blockIdx.x & 15;
    const int mgroup = blockIdx.x >> 4;
    const int n0 = npanel * 64;
    const int m0 = (mgroup * 8 + wslot) * 16;

    const int lm    = lane & 15;
    const int hk    = (lane >> 4) * 8;
    const int mhalf = (lane >> 4) * 8;

    const int srow   = tid >> 2;
    const int schunk = (tid & 3) * 8;
    const _Float16* wsrc = W + (size_t)(n0 + srow) * EMBED + schunk;
    const unsigned  sdst = (unsigned)(uintptr_t)&wtile[srow * WPAD + schunk];

    v8f acc[4] = {};
    const _Float16* xrow = X + (size_t)(m0 + lm) * EMBED;

    for (int k = 0; k < EMBED; k += 32) {
        __syncthreads();
        asm volatile("global_load_async_to_lds_b128 %0, %1, off"
                     :: "v"(sdst), "v"((const void*)(wsrc + k)) : "memory");
        asm volatile("s_wait_asynccnt 0" ::: "memory");
        __syncthreads();

        __builtin_prefetch(xrow + k + 64, 0, 0);
        v8h a0 = *(const v8h*)(xrow + k + hk);
        v8h a1 = *(const v8h*)(xrow + k + hk + 16);
        v16h a;
#pragma unroll
        for (int i = 0; i < 8; ++i) { a[i] = a0[i]; a[8 + i] = a1[i]; }
#pragma unroll
        for (int t = 0; t < 4; ++t) {
            const _Float16* wr = &wtile[(t * 16 + lm) * WPAD + hk];
            v8h c0 = *(const v8h*)(wr);
            v8h c1 = *(const v8h*)(wr + 16);
            v16h b;
#pragma unroll
            for (int i = 0; i < 8; ++i) { b[i] = c0[i]; b[8 + i] = c1[i]; }
            acc[t] = wmma16(a, b, acc[t]);
        }
    }

#pragma unroll
    for (int t = 0; t < 4; ++t)
#pragma unroll
        for (int r = 0; r < 8; ++r) {
            int mrow = m0 + r + mhalf;
            int ncol = n0 + t * 16 + lm;
            Y[(size_t)mrow * EMBED + ncol] = acc[t][r];
        }
}

// ---------------------------------------------------------------------------
extern "C" void kernel_launch(void* const* d_in, const int* in_sizes, int n_in,
                              void* d_out, int out_size, void* d_ws, size_t ws_size,
                              hipStream_t stream) {
    const float* q  = (const float*)d_in[0];
    const float* k  = (const float*)d_in[1];
    const float* v  = (const float*)d_in[2];
    // d_in[3] = causal mask; applied analytically inside attn_kernel
    const float* wq = (const float*)d_in[4];
    const float* wk = (const float*)d_in[5];
    const float* wv = (const float*)d_in[6];
    const float* wo = (const float*)d_in[7];

    char* ws = (char*)d_ws;
    _Float16* wq16 = (_Float16*)(ws + (size_t)(0 << 20));
    _Float16* wk16 = (_Float16*)(ws + (size_t)(2 << 20));
    _Float16* wv16 = (_Float16*)(ws + (size_t)(4 << 20));
    _Float16* wo16 = (_Float16*)(ws + (size_t)(6 << 20));
    _Float16* Qh   = (_Float16*)(ws + (size_t)(8 << 20));   // [B,H,S,D] 16MB
    _Float16* Kh   = (_Float16*)(ws + (size_t)(24 << 20));  // [B,H,S,D] 16MB
    _Float16* VT   = (_Float16*)(ws + (size_t)(40 << 20));  // [B,H,D,S] 16MB
    _Float16* X16  = (_Float16*)(ws + (size_t)(56 << 20));  // [B,S,E]   16MB

    const int nw = EMBED * EMBED;
    cvt_f32_f16<<<(nw + 255) / 256, 256, 0, stream>>>(wq, wq16, nw);
    cvt_f32_f16<<<(nw + 255) / 256, 256, 0, stream>>>(wk, wk16, nw);
    cvt_f32_f16<<<(nw + 255) / 256, 256, 0, stream>>>(wv, wv16, nw);
    cvt_f32_f16<<<(nw + 255) / 256, 256, 0, stream>>>(wo, wo16, nw);

    // 1024 blocks x 8 waves: 64 m-groups x 16 n-panels
    proj_gemm<<<1024, 256, 0, stream>>>(q, wq16, Qh, 0);
    proj_gemm<<<1024, 256, 0, stream>>>(k, wk16, Kh, 0);
    proj_gemm<<<1024, 256, 0, stream>>>(v, wv16, VT, 1);

    // B*H*(S/16) = 8192 waves
    attn_kernel<<<1024, 256, 0, stream>>>(Qh, Kh, VT, X16);

    out_gemm<<<1024, 256, 0, stream>>>(X16, wo16, (float*)d_out);
}
